// GAT_13091060318522
// MI455X (gfx1250) — matrix operand order repaired and checked by reading, not compiled, not run on hardware
//
#include <hip/hip_runtime.h>
#include <math.h>

// ---------- types for WMMA ----------
typedef __attribute__((ext_vector_type(16))) __bf16 v16bf;
typedef __attribute__((ext_vector_type(8)))  __bf16 v8bf;
typedef __attribute__((ext_vector_type(8)))  float  v8f;

// ---------- helpers ----------
static __device__ __forceinline__ unsigned short f32_to_bf16_rne(float f) {
    union { float f; unsigned u; } v; v.f = f;
    unsigned u = v.u;
    unsigned r = u + 0x7FFFu + ((u >> 16) & 1u);
    if ((u & 0x7F800000u) == 0x7F800000u) r = u;   // inf/nan passthrough
    return (unsigned short)(r >> 16);
}

static __device__ __forceinline__ void atomicMaxFloat(float* addr, float val) {
    // signed/unsigned bit trick: valid with -inf init, mixed-sign updates
    if (val >= 0.0f) atomicMax((int*)addr, __float_as_int(val));
    else             atomicMin((unsigned int*)addr, __float_as_uint(val));
}

// ---------- conversion kernels ----------
__global__ void k_f32_to_bf16(const float* __restrict__ in,
                              unsigned short* __restrict__ out, long n) {
    long i = (long)blockIdx.x * blockDim.x + threadIdx.x;
    if (i < n) out[i] = f32_to_bf16_rne(in[i]);
}

// W [128, Fout] row-major f32  ->  Wt [ldc, 128] bf16 (transposed, zero-padded cols)
__global__ void k_w_to_bf16_t(const float* __restrict__ W,
                              unsigned short* __restrict__ Wt,
                              int Fout, int ldc) {
    int i = blockIdx.x * blockDim.x + threadIdx.x;
    if (i >= ldc * 128) return;
    int n = i / 128, k = i % 128;
    float v = (n < Fout) ? W[k * Fout + n] : 0.0f;
    Wt[i] = f32_to_bf16_rne(v);
}

// ---------- WMMA GEMM: C[N,LDC] = A[N,128] (bf16) @ B[128,LDC] (bf16, pre-transposed) ----------
// One 16x16 C tile per wave, K=128 in 4 x v_wmma_f32_16x16x32_bf16.
// REQUIRES N % 16 == 0 (true here: 50000 = 3125*16) -> no row guards, no branches.
template <int LDC>
__global__ void k_gemm_bf16_wmma(const unsigned short* __restrict__ Ab,  // [N,128] bf16
                                 const unsigned short* __restrict__ Wt,  // [LDC,128] bf16
                                 float* __restrict__ C,                  // [N,LDC] f32
                                 int nTilesM, int nTilesN) {
    int tile = blockIdx.x * 8 + (threadIdx.x >> 5);
    if (tile >= nTilesM * nTilesN) return;           // wave-uniform: EXEC all-ones inside
    int tm = tile / nTilesN, tn = tile - tm * nTilesN;
    int m0 = tm * 16, n0 = tn * 16;
    int lane = threadIdx.x & 31;
    int half = lane >> 4;          // 0: lanes 0-15, 1: lanes 16-31
    int l15  = lane & 15;

    const unsigned short* arow = Ab + (m0 + l15) * 128 + half * 16;
    const unsigned short* bcol = Wt + (n0 + l15) * 128 + half * 16;

    v8f c = {0.f, 0.f, 0.f, 0.f, 0.f, 0.f, 0.f, 0.f};
#pragma unroll
    for (int k0 = 0; k0 < 128; k0 += 32) {
        // A frag: half 0 -> K {0..15}, half 1 -> K {16..31} (contiguous, matches B)
        v8bf alo = *reinterpret_cast<const v8bf*>(arow + k0);
        v8bf ahi = *reinterpret_cast<const v8bf*>(arow + k0 + 8);
        v16bf a, b;
#pragma unroll
        for (int i = 0; i < 8; ++i) { a[i] = alo[i]; a[8 + i] = ahi[i]; }
        // B frag: lane holds K = k0 + half*16 + 0..15 for column n0+l15 (contiguous in Wt)
        b = *reinterpret_cast<const v16bf*>(bcol + k0);
        c = __builtin_amdgcn_wmma_f32_16x16x32_bf16(false, a, false, b,
                                                    (short)0, c, false, false);
    }
    // C/D layout: VGPR r -> row m0 + r + half*8, col n0 + l15 ; constant offsets r*LDC
    float* cp = C + (m0 + half * 8) * LDC + n0 + l15;
#pragma unroll
    for (int r = 0; r < 8; ++r) cp[r * LDC] = c[r];
}

// ---------- attention coefficients + softmax-state init ----------
__global__ void k_attn_coef(const float* __restrict__ ft,
                            const float* __restrict__ al, const float* __restrict__ ar,
                            float* __restrict__ el, float* __restrict__ er,
                            float* __restrict__ emax, float* __restrict__ denom,
                            int N, int D, int ldc) {
    int i = blockIdx.x * blockDim.x + threadIdx.x;     // i = n*2 + h
    if (i >= N * 2) return;
    int n = i >> 1, h = i & 1;
    const float* f  = ft + (long)n * ldc + h * D;
    const float* vl = al + h * D;
    const float* vr = ar + h * D;
    float sl = 0.f, sr = 0.f;
    for (int d = 0; d < D; ++d) { float v = f[d]; sl += v * vl[d]; sr += v * vr[d]; }
    el[i] = sl; er[i] = sr;
    emax[i]  = -__builtin_inff();
    denom[i] = 0.f;
}

__global__ void k_zero(float* __restrict__ p, long n) {
    long i = (long)blockIdx.x * blockDim.x + threadIdx.x;
    if (i < n) p[i] = 0.f;
}

// ---------- edge passes ----------
__global__ void k_edge_logits_max(const int* __restrict__ src, const int* __restrict__ dst,
                                  const float* __restrict__ el, const float* __restrict__ er,
                                  float* __restrict__ e, float* __restrict__ emax, int E) {
    int i = blockIdx.x * blockDim.x + threadIdx.x;     // i = eid*2 + h
    if (i >= E * 2) return;
    int eid = i >> 1, h = i & 1;
    int s = src[eid], d = dst[eid];
    float x = el[s * 2 + h] + er[d * 2 + h];
    x = x > 0.f ? x : 0.2f * x;                        // leaky_relu(0.2)
    e[i] = x;
    atomicMaxFloat(&emax[d * 2 + h], x);
}

__global__ void k_edge_exp_sum(const int* __restrict__ dst,
                               float* __restrict__ e, const float* __restrict__ emax,
                               float* __restrict__ denom, int E) {
    int i = blockIdx.x * blockDim.x + threadIdx.x;
    if (i >= E * 2) return;
    int eid = i >> 1, h = i & 1;
    int d = dst[eid];
    float ex = __expf(e[i] - emax[d * 2 + h]);
    e[i] = ex;
    atomicAdd(&denom[d * 2 + h], ex);
}

// one 128-thread block per edge; rst may be pre-initialized with the residual
__global__ void k_edge_aggregate(const int* __restrict__ src, const int* __restrict__ dst,
                                 const float* __restrict__ e, const float* __restrict__ denom,
                                 const float* __restrict__ ft, float* __restrict__ rst,
                                 int Fout, int D, int ldc) {
    int eid = blockIdx.x;
    int s = src[eid], d = dst[eid];
    float a0 = e[eid * 2 + 0] / denom[d * 2 + 0];
    float a1 = e[eid * 2 + 1] / denom[d * 2 + 1];
    const float* fs = ft + (long)s * ldc;
    float*       rd = rst + (long)d * ldc;
    for (int c = threadIdx.x; c < Fout; c += blockDim.x) {
        float a = (c < D) ? a0 : a1;
        atomicAdd(&rd[c], fs[c] * a);
    }
}

// ---------- epilogues ----------
// rst[n,c] = ELU(rst + bias[c] (+ res)) ; ldc == F (layers 0/1)
__global__ void k_bias_res_elu(float* __restrict__ rst, const float* __restrict__ bias,
                               const float* __restrict__ res, long total, int F) {
    long i = (long)blockIdx.x * blockDim.x + threadIdx.x;
    if (i >= total) return;
    int c = (int)(i % F);
    float v = rst[i] + bias[c];
    if (res) v += res[i];
    rst[i] = v > 0.f ? v : (__expf(v) - 1.f);
}

// out[n,c] = mean over heads of (rst + b2), n < N-1 ; rst layout [N,208], heads at 0..96 / 97..193
__global__ void k_final_mean(const float* __restrict__ rst, const float* __restrict__ b2,
                             float* __restrict__ out, int Nout) {
    int i = blockIdx.x * blockDim.x + threadIdx.x;
    if (i >= Nout * 97) return;
    int n = i / 97, c = i % 97;
    const float* r = rst + (long)n * 208;
    out[i] = 0.5f * ((r[c] + b2[c]) + (r[97 + c] + b2[97 + c]));
}

// ---------- host ----------
static inline int cdiv(long a, int b) { return (int)((a + b - 1) / b); }

extern "C" void kernel_launch(void* const* d_in, const int* in_sizes, int n_in,
                              void* d_out, int out_size, void* d_ws, size_t ws_size,
                              hipStream_t stream) {
    const float* x     = (const float*)d_in[0];
    const int*   esrc  = (const int*)  d_in[1];
    const int*   edst  = (const int*)  d_in[2];
    const float* W0    = (const float*)d_in[3];
    const float* al0   = (const float*)d_in[4];
    const float* ar0   = (const float*)d_in[5];
    const float* b0    = (const float*)d_in[6];
    const float* W1    = (const float*)d_in[7];
    const float* al1   = (const float*)d_in[8];
    const float* ar1   = (const float*)d_in[9];
    const float* b1    = (const float*)d_in[10];
    const float* W2    = (const float*)d_in[11];
    const float* al2   = (const float*)d_in[12];
    const float* ar2   = (const float*)d_in[13];
    const float* b2    = (const float*)d_in[14];
    const float* resW2 = (const float*)d_in[15];

    const int N = in_sizes[0] / 128;   // 50000 (divisible by 16)
    const int E = in_sizes[1];         // 800000

    // workspace carve (256B aligned); total ~146 MB
    char* w = (char*)d_ws;
    auto carve = [&](size_t bytes) { void* p = (void*)w; w += (bytes + 255) & ~(size_t)255; return p; };
    float*          bufA  = (float*)carve((size_t)N * 208 * 4);   // ft (padded)
    float*          bufB  = (float*)carve((size_t)N * 208 * 4);   // rst0 / res2+rst2
    float*          bufC  = (float*)carve((size_t)N * 208 * 4);   // rst1
    unsigned short* hb    = (unsigned short*)carve((size_t)N * 128 * 2);
    unsigned short* Wt    = (unsigned short*)carve((size_t)208 * 128 * 2);
    float*          elb   = (float*)carve((size_t)N * 2 * 4);
    float*          erb   = (float*)carve((size_t)N * 2 * 4);
    float*          emaxb = (float*)carve((size_t)N * 2 * 4);
    float*          denb  = (float*)carve((size_t)N * 2 * 4);
    float*          ebuf  = (float*)carve((size_t)E * 2 * 4);

    const int nTilesM = N / 16;        // 3125 (exact)

    auto gemm = [&](const unsigned short* A, const unsigned short* B, float* C, int ldc) {
        int nTilesN = ldc / 16;
        int tiles = nTilesM * nTilesN;
        if (ldc == 128)
            k_gemm_bf16_wmma<128><<<cdiv(tiles, 8), 256, 0, stream>>>(A, B, C, nTilesM, nTilesN);
        else
            k_gemm_bf16_wmma<208><<<cdiv(tiles, 8), 256, 0, stream>>>(A, B, C, nTilesM, nTilesN);
    };
    auto edges = [&](const float* ft, float* rst, const float* al, const float* ar,
                     int Fout, int D, int ldc) {
        k_attn_coef<<<cdiv((long)N * 2, 256), 256, 0, stream>>>(ft, al, ar, elb, erb, emaxb, denb, N, D, ldc);
        k_edge_logits_max<<<cdiv((long)E * 2, 256), 256, 0, stream>>>(esrc, edst, elb, erb, ebuf, emaxb, E);
        k_edge_exp_sum<<<cdiv((long)E * 2, 256), 256, 0, stream>>>(edst, ebuf, emaxb, denb, E);
        k_edge_aggregate<<<E, 128, 0, stream>>>(esrc, edst, ebuf, denb, ft, rst, Fout, D, ldc);
    };

    // ---------------- layer 0: x -> h1 (bufB), no residual, ELU ----------------
    k_f32_to_bf16<<<cdiv((long)N * 128, 256), 256, 0, stream>>>(x, hb, (long)N * 128);
    k_w_to_bf16_t<<<cdiv(128 * 128, 256), 256, 0, stream>>>(W0, Wt, 128, 128);
    gemm(hb, Wt, bufA, 128);
    k_zero<<<cdiv((long)N * 128, 256), 256, 0, stream>>>(bufB, (long)N * 128);
    edges(bufA, bufB, al0, ar0, 128, 64, 128);
    k_bias_res_elu<<<cdiv((long)N * 128, 256), 256, 0, stream>>>(bufB, b0, nullptr, (long)N * 128, 128);

    // ---------------- layer 1: h1 -> h2 (bufC), identity residual, ELU ----------------
    k_f32_to_bf16<<<cdiv((long)N * 128, 256), 256, 0, stream>>>(bufB, hb, (long)N * 128);
    k_w_to_bf16_t<<<cdiv(128 * 128, 256), 256, 0, stream>>>(W1, Wt, 128, 128);
    gemm(hb, Wt, bufA, 128);
    k_zero<<<cdiv((long)N * 128, 256), 256, 0, stream>>>(bufC, (long)N * 128);
    edges(bufA, bufC, al1, ar1, 128, 64, 128);
    k_bias_res_elu<<<cdiv((long)N * 128, 256), 256, 0, stream>>>(bufC, b1, bufB, (long)N * 128, 128);

    // ---------------- layer 2: h2 -> logits, linear residual, mean heads ----------------
    k_f32_to_bf16<<<cdiv((long)N * 128, 256), 256, 0, stream>>>(bufC, hb, (long)N * 128);
    k_w_to_bf16_t<<<cdiv(208 * 128, 256), 256, 0, stream>>>(W2, Wt, 194, 208);
    gemm(hb, Wt, bufA, 208);                         // ft2 [N,208]
    k_w_to_bf16_t<<<cdiv(208 * 128, 256), 256, 0, stream>>>(resW2, Wt, 194, 208);
    gemm(hb, Wt, bufB, 208);                         // res2 [N,208] (= rst init)
    edges(bufA, bufB, al2, ar2, 194, 97, 208);       // scatter accumulates onto res2
    k_final_mean<<<cdiv((long)(N - 1) * 97, 256), 256, 0, stream>>>(bufB, b2, (float*)d_out, N - 1);
}
